// CustomRNN_72224170049939
// MI455X (gfx1250) — matrix operand order repaired and checked by reading, not compiled, not run on hardware
//
#include <hip/hip_runtime.h>
#include <hip/hip_bf16.h>

typedef __bf16 bf16_t;
typedef __attribute__((ext_vector_type(16))) __bf16 v16bf;
typedef __attribute__((ext_vector_type(8)))  __bf16 v8bf;
typedef __attribute__((ext_vector_type(8)))  float  v8f;

#define TDIM 2048
#define BDIM 64
#define XDIM 256
#define YDIM 64
#define UDIM 32
#define YHD  64
#define ZHD  16

// ---------------------------------------------------------------------------
// WMMA helper (gfx1250: D = A(16x32) * B(32x16) + C, wave32)
// ---------------------------------------------------------------------------
static __device__ __forceinline__ v8f wmma_bf16(v16bf a, v16bf b, v8f c) {
  return __builtin_amdgcn_wmma_f32_16x16x32_bf16(false, a, false, b,
                                                 (short)0, c, false, false);
}

// A-fragment (16x32 bf16) from f32 row-major source, per ISA VGPR layout:
// lane L (m=L&15, hi=L>>4): j=0..7 -> k=k0+j ; j=8..15 -> k=k0+16+(j-8),
// k0 = kt*32 + hi*8. `p` points at &src[m][k0].
static __device__ __forceinline__ v16bf afrag_f32(const float* p) {
  v16bf a;
#pragma unroll
  for (int j = 0; j < 8; ++j) a[j] = (bf16_t)p[j];
#pragma unroll
  for (int j = 0; j < 8; ++j) a[j + 8] = (bf16_t)p[16 + j];
  return a;
}

// Same from a bf16 row-major source (two 16B vector loads, no conversion).
static __device__ __forceinline__ v16bf afrag_bf16(const bf16_t* p) {
  v8bf lo = *(const v8bf*)p;
  v8bf hi = *(const v8bf*)(p + 16);
  v16bf a;
#pragma unroll
  for (int j = 0; j < 8; ++j) { a[j] = lo[j]; a[j + 8] = hi[j]; }
  return a;
}

// ---------------------------------------------------------------------------
// CDNA5 async LDS<->global (ASYNCcnt-tracked, no VGPR roundtrip)
// ---------------------------------------------------------------------------
static __device__ __forceinline__ void async_store_b128(const bf16_t* lds_src,
                                                        bf16_t* gdst) {
  unsigned lds = (unsigned)(uintptr_t)lds_src;  // low 32b of flat = LDS offset
  asm volatile("global_store_async_from_lds_b128 %0, %1, off"
               :: "v"(gdst), "v"(lds) : "memory");
}
static __device__ __forceinline__ void async_load_b128(bf16_t* lds_dst,
                                                       const bf16_t* gsrc) {
  unsigned lds = (unsigned)(uintptr_t)lds_dst;
  asm volatile("global_load_async_to_lds_b128 %0, %1, off"
               :: "v"(lds), "v"(gsrc) : "memory");
}
static __device__ __forceinline__ void wait_async_le1() {
  asm volatile("s_wait_asynccnt 0x1" ::: "memory");
}
static __device__ __forceinline__ void wait_async_0() {
  asm volatile("s_wait_asynccnt 0x0" ::: "memory");
}

// ---------------------------------------------------------------------------
// Kernel 0: pack f32 weight W[N][K] (used as W^T) into bf16 B-fragment order:
// dst[((nt*Kt+kt)*32+lane)*16+j] = W[nt*16+(lane&15)][kt*32+(lane>>4)*8+(j&7)+(j>>3)*16]
// ---------------------------------------------------------------------------
__global__ void pack_w(const float* __restrict__ src, bf16_t* __restrict__ dst,
                       int N, int K) {
  int idx = blockIdx.x * 256 + threadIdx.x;
  int Kt = K >> 5;
  int total = (N >> 4) * Kt * 512;
  if (idx >= total) return;
  int j    = idx & 15;
  int lane = (idx >> 4) & 31;
  int tile = idx >> 9;
  int kt   = tile % Kt;
  int nt   = tile / Kt;
  int n = nt * 16 + (lane & 15);
  int k = kt * 32 + (lane >> 4) * 8 + (j & 7) + (j >> 3) * 16;
  dst[idx] = (bf16_t)src[(size_t)n * K + k];
}

// ---------------------------------------------------------------------------
// Kernel 1 (parallel): drive in C-fragment layout (f32):
// drive[(((t*4+rb)*16+nt)*256) + lane*8 + r] =
//   A_b[n] + K_b[n] + Bm_b[n] + (y_t Kw^T)[m,n] + (u_t Bm^T)[m,n]
// ---------------------------------------------------------------------------
__global__ __launch_bounds__(256) void drive_kernel(
    const float* __restrict__ y, const float* __restrict__ u,
    const float* __restrict__ A_b, const float* __restrict__ K_b,
    const float* __restrict__ Bm_b,
    const bf16_t* __restrict__ pkK, const bf16_t* __restrict__ pkBm,
    float* __restrict__ drive) {
  const int lane = threadIdx.x & 31;
  const int wv   = threadIdx.x >> 5;
  const size_t tile = (size_t)blockIdx.x * 8 + wv;   // (t*4+rb)*16+nt
  const int nt = (int)(tile & 15);
  const int rb = (int)((tile >> 4) & 3);
  const int t  = (int)(tile >> 6);
  const int m  = lane & 15;
  const int hi = lane >> 4;
  const int n  = nt * 16 + m;
  const int rowbase = rb * 16;

  v8f c;
  const float b0 = A_b[n] + K_b[n] + Bm_b[n];
#pragma unroll
  for (int r = 0; r < 8; ++r) c[r] = b0;

#pragma unroll
  for (int kt = 0; kt < 2; ++kt) {
    const float* ap =
        y + ((size_t)(rowbase + m) * TDIM + t) * YDIM + kt * 32 + hi * 8;
    v16bf b = *(const v16bf*)(pkK + ((nt * 2 + kt) * 32 + lane) * 16);
    c = wmma_bf16(afrag_f32(ap), b, c);
  }
  {
    const float* ap = u + ((size_t)(rowbase + m) * TDIM + t) * UDIM + hi * 8;
    v16bf b = *(const v16bf*)(pkBm + (nt * 32 + lane) * 16);
    c = wmma_bf16(afrag_f32(ap), b, c);
  }
  *(v8f*)(drive + tile * 256 + lane * 8) = c;
}

// ---------------------------------------------------------------------------
// Kernel 2 (sequential scan): 4 blocks x 512 threads (16 waves).
// Block rb owns batch rows [rb*16, rb*16+16); wave w owns col tile nt=w.
// bf16 state double-buffered in LDS, A_w fragments register-resident,
// drive fragment software-pipelined, dual WMMA accumulator chains,
// x_t history streamed out with async LDS->global stores.
// ---------------------------------------------------------------------------
__global__ __launch_bounds__(512) void recurrence_kernel(
    const float* __restrict__ drive, const bf16_t* __restrict__ pkA,
    bf16_t* __restrict__ states) {
  __shared__ bf16_t xbuf[2][16][XDIM];

  const int tid  = threadIdx.x;
  const int rb   = blockIdx.x;
  const int lane = tid & 31;
  const int w    = tid >> 5;
  const int m    = lane & 15;
  const int hi   = lane >> 4;
  const int n    = w * 16 + m;
  const int sr   = tid >> 5;          // streaming row for this thread
  const int sc   = (tid & 31) * 8;    // streaming col0 for this thread

  // x_0 = 0 (bf16 zeros), one 16B chunk per thread
  {
    v8bf z = {};
    *(v8bf*)(&xbuf[0][sr][sc]) = z;
  }
  __syncthreads();

  // A_w B-fragments live in VGPRs for the whole scan (8 x v16bf = 64 VGPRs)
  v16bf bA[8];
#pragma unroll
  for (int kt = 0; kt < 8; ++kt)
    bA[kt] = *(const v16bf*)(pkA + ((w * 8 + kt) * 32 + lane) * 16);

  // loop-invariant zero C operand for the second accumulator chain
  v8f zc = {};

  // running pointers (step strides: drive +16384 f32, states +256 bf16)
  const float* dptr =
      drive + ((size_t)rb * 16 + w) * 256 + lane * 8;       // t = 0
  bf16_t* sptr = states + (size_t)(rb * 16 + sr) * TDIM * XDIM + sc;

  v8f dcur = *(const v8f*)dptr;   // drive fragment for t = 0

  int cur = 0;
  for (int t = 0; t < TDIM; ++t) {
    // stream x_t (pre-update) to global, LDS -> memory, ASYNCcnt-tracked
    async_store_b128(&xbuf[cur][sr][sc], sptr);
    sptr += XDIM;

    // prefetch next step's drive fragment (t=TDIM-1 reads one tile past the
    // drive region -- still inside the workspace, value never used)
    dptr += 4 * 16 * 256;
    v8f dnxt = *(const v8f*)dptr;

    // x_t @ A^T : 8 K-tiles, two independent accumulator chains
    v8f c0 = dcur;
    v8f c1 = zc;
#pragma unroll
    for (int kt = 0; kt < 8; kt += 2) {
      c0 = wmma_bf16(afrag_bf16(&xbuf[cur][m][kt * 32 + hi * 8]), bA[kt], c0);
      c1 = wmma_bf16(afrag_bf16(&xbuf[cur][m][(kt + 1) * 32 + hi * 8]),
                     bA[kt + 1], c1);
    }
    v8f c = c0 + c1;

    // previous step's async store must be done before we overwrite its buffer
    wait_async_le1();

    const int nxt = cur ^ 1;
#pragma unroll
    for (int r = 0; r < 8; ++r)
      xbuf[nxt][r + 8 * hi][n] = (bf16_t)c[r];
    __syncthreads();
    cur = nxt;
    dcur = dnxt;
  }
}

// ---------------------------------------------------------------------------
// Kernel 3 (parallel): heads. One block (16 waves) per 16-row tile of the
// flattened (b*T+t) axis. State tile async-staged into LDS once per block.
// ---------------------------------------------------------------------------
__global__ __launch_bounds__(512) void heads_kernel(
    const bf16_t* __restrict__ states,
    const float* __restrict__ Cy1_b, const float* __restrict__ Cz1_b,
    const float* __restrict__ Cy2_b, const float* __restrict__ Cz2_b,
    const bf16_t* __restrict__ pkCy1, const bf16_t* __restrict__ pkCz1,
    const bf16_t* __restrict__ pkCy2, const bf16_t* __restrict__ pkCz2,
    float* __restrict__ outy, float* __restrict__ outz) {
  __shared__ bf16_t xs[16][XDIM];        // staged state tile (8 KB)
  __shared__ bf16_t h[2][16][XDIM];      // relu hidden, both heads (16 KB)

  const int tid  = threadIdx.x;
  const int lane = tid & 31;
  const int w    = tid >> 5;
  const int m    = lane & 15;
  const int hi   = lane >> 4;
  const size_t row0 = (size_t)blockIdx.x * 16;

  // stage the 16x256 bf16 state tile: one async 16B chunk per thread
  {
    int r = tid >> 5, c0 = (tid & 31) * 8;
    async_load_b128(&xs[r][c0], states + (row0 + r) * XDIM + c0);
    wait_async_0();
  }
  __syncthreads();

  // ---- stage 1: h = relu(x C1^T + b1) for both heads
#pragma unroll
  for (int head = 0; head < 2; ++head) {
    const bf16_t* pk = head ? pkCz1 : pkCy1;
    const float* bb  = head ? Cz1_b : Cy1_b;
    const int n = w * 16 + m;
    v8f c;
    const float b0 = bb[n];
#pragma unroll
    for (int r = 0; r < 8; ++r) c[r] = b0;
#pragma unroll
    for (int kt = 0; kt < 8; ++kt) {
      v16bf b = *(const v16bf*)(pk + ((w * 8 + kt) * 32 + lane) * 16);
      c = wmma_bf16(afrag_bf16(&xs[m][kt * 32 + hi * 8]), b, c);
    }
#pragma unroll
    for (int r = 0; r < 8; ++r) {
      float v = c[r];
      h[head][r + 8 * hi][n] = (bf16_t)(v > 0.0f ? v : 0.0f);
    }
  }
  __syncthreads();

  // ---- stage 2: sigmoid(h @ C2^T + b2); wave-uniform branch keeps EXEC full
  if (w < 5) {
    const bool isz    = (w == 4);
    const bf16_t* pk2 = isz ? pkCz2 : pkCy2;
    const float* b2   = isz ? Cz2_b : Cy2_b;
    const int nt = isz ? 0 : w;
    const int n  = nt * 16 + m;
    v8f c;
    const float b0 = b2[n];
#pragma unroll
    for (int r = 0; r < 8; ++r) c[r] = b0;
#pragma unroll
    for (int kt = 0; kt < 8; ++kt) {
      v16bf b = *(const v16bf*)(pk2 + ((nt * 8 + kt) * 32 + lane) * 16);
      c = wmma_bf16(afrag_bf16(&h[isz ? 1 : 0][m][kt * 32 + hi * 8]), b, c);
    }
#pragma unroll
    for (int r = 0; r < 8; ++r) {
      size_t row = row0 + r + 8 * hi;
      float v = 1.0f / (1.0f + __expf(-c[r]));
      if (isz) outz[row * ZHD + n] = v;
      else     outy[row * YHD + n] = v;
    }
  }
}

// ---------------------------------------------------------------------------
// Workspace layout (bytes)
// ---------------------------------------------------------------------------
#define OFF_A     0u            // 128 KB
#define OFF_K     131072u       //  32 KB
#define OFF_BM    163840u       //  16 KB
#define OFF_CY1   180224u       // 128 KB
#define OFF_CZ1   311296u       // 128 KB
#define OFF_CY2   442368u       //  32 KB
#define OFF_CZ2   475136u       //   8 KB
#define OFF_DRV   524288u       // drive f32, B*T*X = 128 MB
#define OFF_ST    134742016u    // states bf16, B*T*X = 64 MB

extern "C" void kernel_launch(void* const* d_in, const int* in_sizes, int n_in,
                              void* d_out, int out_size, void* d_ws,
                              size_t ws_size, hipStream_t stream) {
  (void)in_sizes; (void)n_in; (void)out_size; (void)ws_size;
  const float* y     = (const float*)d_in[0];
  const float* u     = (const float*)d_in[1];
  const float* A_w   = (const float*)d_in[2];
  const float* A_b   = (const float*)d_in[3];
  const float* K_w   = (const float*)d_in[4];
  const float* K_b   = (const float*)d_in[5];
  const float* Bm_w  = (const float*)d_in[6];
  const float* Bm_b  = (const float*)d_in[7];
  const float* Cy1_w = (const float*)d_in[8];
  const float* Cy1_b = (const float*)d_in[9];
  const float* Cy2_w = (const float*)d_in[10];
  const float* Cy2_b = (const float*)d_in[11];
  const float* Cz1_w = (const float*)d_in[12];
  const float* Cz1_b = (const float*)d_in[13];
  const float* Cz2_w = (const float*)d_in[14];
  const float* Cz2_b = (const float*)d_in[15];

  char* ws = (char*)d_ws;
  bf16_t* pkA    = (bf16_t*)(ws + OFF_A);
  bf16_t* pkK    = (bf16_t*)(ws + OFF_K);
  bf16_t* pkBm   = (bf16_t*)(ws + OFF_BM);
  bf16_t* pkCy1  = (bf16_t*)(ws + OFF_CY1);
  bf16_t* pkCz1  = (bf16_t*)(ws + OFF_CZ1);
  bf16_t* pkCy2  = (bf16_t*)(ws + OFF_CY2);
  bf16_t* pkCz2  = (bf16_t*)(ws + OFF_CZ2);
  float*  drive  = (float*) (ws + OFF_DRV);
  bf16_t* states = (bf16_t*)(ws + OFF_ST);

  float* outy = (float*)d_out;
  float* outz = outy + (size_t)BDIM * TDIM * YHD;

  // pack all weights into WMMA B-fragment order (bf16)
  pack_w<<<256, 256, 0, stream>>>(A_w,   pkA,   XDIM, XDIM);
  pack_w<<< 64, 256, 0, stream>>>(K_w,   pkK,   XDIM, YDIM);
  pack_w<<< 32, 256, 0, stream>>>(Bm_w,  pkBm,  XDIM, UDIM);
  pack_w<<<256, 256, 0, stream>>>(Cy1_w, pkCy1, XDIM, XDIM);
  pack_w<<<256, 256, 0, stream>>>(Cz1_w, pkCz1, XDIM, XDIM);
  pack_w<<< 64, 256, 0, stream>>>(Cy2_w, pkCy2, YHD,  XDIM);
  pack_w<<< 16, 256, 0, stream>>>(Cz2_w, pkCz2, ZHD,  XDIM);

  // parallel: drive in C-fragment layout (f32)
  drive_kernel<<<(TDIM * 4 * 16) / 8, 256, 0, stream>>>(
      y, u, A_b, K_b, Bm_b, pkK, pkBm, drive);

  // sequential scan over T (lean critical path, async state streaming)
  recurrence_kernel<<<4, 512, 0, stream>>>(drive, pkA, states);

  // parallel heads over all B*T states
  heads_kernel<<<(BDIM * TDIM) / 16, 512, 0, stream>>>(
      states, Cy1_b, Cz1_b, Cy2_b, Cz2_b, pkCy1, pkCz1, pkCy2, pkCz2,
      outy, outz);
}
